// SimplePRMoEModel_84920093376586
// MI455X (gfx1250) — compile-verified, matrix operand order checked by reading, not run on hardware
//
#include <hip/hip_runtime.h>
#include <hip/hip_bf16.h>

typedef __attribute__((ext_vector_type(16))) __bf16 v16bf;
typedef __attribute__((ext_vector_type(8)))  float  v8f;
typedef __attribute__((__vector_size__(4 * sizeof(int)))) int vi4;

#define GAS __attribute__((address_space(1)))
#define LAS __attribute__((address_space(3)))

#if defined(__has_builtin)
#if __has_builtin(__builtin_amdgcn_global_load_async_to_lds_b128)
#define USE_ASYNC_A 1
#else
#define USE_ASYNC_A 0
#endif
#else
#define USE_ASYNC_A 0
#endif

// Problem dims (fixed by the reference setup)
constexpr int NB   = 8;
constexpr int SEQ  = 2048;
constexpr int TTOK = NB * SEQ;   // 16384 tokens
constexpr int DIM  = 1024;
constexpr int DFF  = 4096;
constexpr int E1   = 8;
constexpr int E2   = 16;
constexpr int CAP1 = 2560;       // 1.25*T/E1
constexpr int CAP2 = 1280;       // 1.25*T/E2
constexpr int ROWS = E1 * CAP1;  // == E2*CAP2 == 20480 (shared scratch rows)

#if USE_ASYNC_A
__device__ __forceinline__ void wait_async0() {
#if __has_builtin(__builtin_amdgcn_s_wait_asynccnt)
  __builtin_amdgcn_s_wait_asynccnt(0);
#else
  asm volatile("s_wait_asynccnt 0" ::: "memory");
#endif
}
#endif

// ---------------------------------------------------------------------------
// Gating: one wave per token. Computes logits x@wg, softmax top-1 index+prob.
// ---------------------------------------------------------------------------
__global__ __launch_bounds__(256) void gate_kernel(const float* __restrict__ xin,
                                                   const float* __restrict__ wg,
                                                   int E,
                                                   int* __restrict__ idx,
                                                   float* __restrict__ gate) {
  const int wave = threadIdx.x >> 5;
  const int lane = threadIdx.x & 31;
  const int t = blockIdx.x * 8 + wave;
  float acc[16];
  #pragma unroll
  for (int e = 0; e < 16; ++e) acc[e] = 0.f;
  const float* xr = xin + (size_t)t * DIM;
  for (int c = lane; c < DIM; c += 32) {
    const float xv = xr[c];
    const float* wr = wg + c * E;
    for (int e = 0; e < E; ++e) acc[e] += xv * wr[e];
  }
  for (int e = 0; e < E; ++e)
    for (int off = 16; off > 0; off >>= 1)
      acc[e] += __shfl_xor(acc[e], off, 32);
  if (lane == 0) {
    float mx = acc[0]; int am = 0;
    for (int e = 1; e < E; ++e) if (acc[e] > mx) { mx = acc[e]; am = e; }
    float s = 0.f;
    for (int e = 0; e < E; ++e) s += __expf(acc[e] - mx);
    idx[t] = am;
    gate[t] = 1.f / s;           // max softmax prob
  }
}

// ---------------------------------------------------------------------------
// Deterministic per-expert positions (token-order prefix count, like cumsum).
// ---------------------------------------------------------------------------
__global__ void pos_kernel(const int* __restrict__ idx, int* __restrict__ pos) {
  const int e = blockIdx.x;
  const int lane = threadIdx.x;  // 32 threads
  int base = 0;
  for (int t0 = 0; t0 < TTOK; t0 += 32) {
    const int t = t0 + lane;
    const bool match = (idx[t] == e);
    const unsigned mask = (unsigned)__ballot(match);
    if (match) pos[t] = base + __popc(mask & ((1u << lane) - 1u));
    base += __popc(mask);
  }
}

__global__ void fill_zero4(uint4* __restrict__ p, int n4) {
  const int i = blockIdx.x * blockDim.x + threadIdx.x;
  uint4 z; z.x = 0u; z.y = 0u; z.z = 0u; z.w = 0u;
  if (i < n4) p[i] = z;
}

// Gather kept tokens into per-expert bf16 buffers buf[E][cap][DIM].
__global__ __launch_bounds__(256) void gather_kernel(const float* __restrict__ xin,
                                                     const int* __restrict__ idx,
                                                     const int* __restrict__ pos,
                                                     int cap,
                                                     __bf16* __restrict__ buf) {
  const int t = blockIdx.x;
  const int p = pos[t];
  if (p >= cap) return;
  const size_t row = (size_t)idx[t] * cap + p;
  const int c4 = threadIdx.x;  // DIM/4 == 256
  const float4 xv = ((const float4*)(xin + (size_t)t * DIM))[c4];
  union { __bf16 h[4]; uint2 u; } pk;
  pk.h[0] = (__bf16)xv.x; pk.h[1] = (__bf16)xv.y;
  pk.h[2] = (__bf16)xv.z; pk.h[3] = (__bf16)xv.w;
  ((uint2*)(buf + row * DIM))[c4] = pk.u;
}

// ---------------------------------------------------------------------------
// bf16 WMMA GEMM: C[e] = act( A[e](bf16 MxK) * Bw[e](f32 KxN -> bf16) + bias )
// Block tile 128x128, 8 waves (4x2), each wave 32x64 = 2x4 WMMA tiles.
// LDS double-buffered. A tile staged with async global->LDS b128 DMA
// (ASYNCcnt path); B staged fragment-ready as packed bf16 K-pairs so
// fragment reads are plain ds_load_b128.
// ---------------------------------------------------------------------------
template <bool RELU, bool OUT_BF16>
__global__ __launch_bounds__(256) void gemm_bf16_wmma(
    const __bf16* __restrict__ A, const float* __restrict__ Bw,
    const float* __restrict__ bias, void* __restrict__ Cout,
    int K, int N,
    long long strA, long long strB, long long strBias, long long strC) {
  // sA: 128 rows x 32 bf16 (16 uints per row).  sB: frag-ready, (n, half, i).
  __shared__ unsigned sAu[2][128 * 16];
  __shared__ unsigned sBu[2][128 * 2 * 8];

  const int e  = blockIdx.z;
  const int m0 = blockIdx.y * 128;
  const int n0 = blockIdx.x * 128;
  const __bf16* Ae = A + (size_t)e * strA + (size_t)m0 * K;
  const float*  Be = Bw + (size_t)e * strB + n0;
  const float*  be = bias + (size_t)e * strBias + n0;
  char*         Ce = (char*)Cout + (size_t)e * strC * (OUT_BF16 ? 2 : 4);

  const int tid  = threadIdx.x;
  const int lane = tid & 31;
  const int wave = tid >> 5;
  const int wm   = wave & 3;   // 4 M groups of 32 rows
  const int wn   = wave >> 2;  // 2 N groups of 64 cols
  const int half = lane >> 4;
  const int l16  = lane & 15;

  // Staging roles (fixed per thread)
  const int arow0 = tid >> 2;        // rows tid/4 and tid/4 + 64
  const int aq    = tid & 3;         // 8-bf16 segment within row
  const int bn    = tid & 127;       // B column within tile
  const int bh    = tid >> 7;        // B half (K-halves 0..7 / 8..15 pattern)

  auto stage = [&](int kt, int b) {
    // ---- A tile: 128x32 bf16, two b128 transfers per thread
#if USE_ASYNC_A
    __builtin_amdgcn_global_load_async_to_lds_b128(
        (GAS vi4*)(Ae + arow0 * K + kt + aq * 8),
        (LAS vi4*)&sAu[b][arow0 * 16 + aq * 4],
        0, 0);
    __builtin_amdgcn_global_load_async_to_lds_b128(
        (GAS vi4*)(Ae + (arow0 + 64) * K + kt + aq * 8),
        (LAS vi4*)&sAu[b][(arow0 + 64) * 16 + aq * 4],
        0, 0);
#else
    uint4 a0 = *(const uint4*)(Ae + arow0 * K + kt + aq * 8);
    uint4 a1 = *(const uint4*)(Ae + (arow0 + 64) * K + kt + aq * 8);
#endif
    // ---- B tile: 32x128 f32 -> packed bf16 pairs in ISA k-order
    unsigned bu[8];
    #pragma unroll
    for (int i = 0; i < 8; ++i) {
      const int kl = ((i < 4) ? (2 * i) : (16 + 2 * (i - 4))) + 8 * bh;
      const float f0 = Be[(kt + kl) * N + bn];
      const float f1 = Be[(kt + kl + 1) * N + bn];
      union { __bf16 h[2]; unsigned u; } pk;
      pk.h[0] = (__bf16)f0; pk.h[1] = (__bf16)f1;
      bu[i] = pk.u;
    }
    // Prefetch the weight stream two K-tiles ahead (global_prefetch_b8)
    if (kt + 96 <= K) __builtin_prefetch(&Be[(kt + 64) * N + bn], 0, 1);
#if !USE_ASYNC_A
    *(uint4*)&sAu[b][arow0 * 16 + aq * 4] = a0;
    *(uint4*)&sAu[b][(arow0 + 64) * 16 + aq * 4] = a1;
#endif
    unsigned* bd = &sBu[b][(bn * 2 + bh) * 8];
    uint4 q0; q0.x = bu[0]; q0.y = bu[1]; q0.z = bu[2]; q0.w = bu[3];
    uint4 q1; q1.x = bu[4]; q1.y = bu[5]; q1.z = bu[6]; q1.w = bu[7];
    *(uint4*)&bd[0] = q0;
    *(uint4*)&bd[4] = q1;
  };

  v8f acc[2][4] = {};
  union Frag { uint4 q[2]; v16bf v; };

  int cur = 0;
  stage(0, cur);
  for (int kt = 0; kt < K; kt += 32) {
#if USE_ASYNC_A
    wait_async0();      // our async LDS writes from the previous stage() done
#endif
    __syncthreads();    // publish tile 'cur' to all waves
    const int nxt = cur ^ 1;
    if (kt + 32 < K) stage(kt + 32, nxt);

    // A fragments: u[0..3]=row[4*half+0..3], u[4..7]=row[8+4*half+0..3]
    Frag af[2];
    #pragma unroll
    for (int tm = 0; tm < 2; ++tm) {
      const int m = wm * 32 + tm * 16 + l16;
      const unsigned* ra = &sAu[cur][m * 16];
      af[tm].q[0] = *(const uint4*)&ra[4 * half];
      af[tm].q[1] = *(const uint4*)&ra[8 + 4 * half];
    }
    // B fragments: contiguous packed pairs at (n, half)
    Frag bfr[4];
    #pragma unroll
    for (int tn = 0; tn < 4; ++tn) {
      const int n = wn * 64 + tn * 16 + l16;
      const unsigned* rb = &sBu[cur][(n * 2 + half) * 8];
      bfr[tn].q[0] = *(const uint4*)&rb[0];
      bfr[tn].q[1] = *(const uint4*)&rb[4];
    }
    #pragma unroll
    for (int tm = 0; tm < 2; ++tm)
      #pragma unroll
      for (int tn = 0; tn < 4; ++tn)
        acc[tm][tn] = __builtin_amdgcn_wmma_f32_16x16x32_bf16(
            false, af[tm].v, false, bfr[tn].v, (short)0, acc[tm][tn], false, false);
    cur = nxt;
  }

  // Epilogue: C layout: VGPR r -> M=r (+8 upper half), N = lane%16
  #pragma unroll
  for (int tm = 0; tm < 2; ++tm) {
    #pragma unroll
    for (int tn = 0; tn < 4; ++tn) {
      const int nloc = wn * 64 + tn * 16 + l16;
      const int n = n0 + nloc;
      const float bv = be[nloc];
      #pragma unroll
      for (int r = 0; r < 8; ++r) {
        const int m = m0 + wm * 32 + tm * 16 + r + half * 8;
        float v = acc[tm][tn][r] + bv;
        if (RELU) v = fmaxf(v, 0.f);
        if (OUT_BF16)
          ((__bf16*)Ce)[(size_t)m * N + n] = (__bf16)v;
        else
          ((float*)Ce)[(size_t)m * N + n] = v;
      }
    }
  }
}

// Gated scatter back to token order; optionally fuse residual add (float4 path).
__global__ __launch_bounds__(256) void scatter_kernel(const float* __restrict__ eo,
                                                      const int* __restrict__ idx,
                                                      const int* __restrict__ pos,
                                                      const float* __restrict__ gate,
                                                      int cap,
                                                      const float* __restrict__ residAdd,
                                                      float* __restrict__ out) {
  const int t = blockIdx.x;
  const int p = pos[t];
  const float g = (p < cap) ? gate[t] : 0.f;
  const int pc = (p < cap) ? p : (cap - 1);
  const size_t row = (size_t)idx[t] * cap + pc;
  const int c4 = threadIdx.x;  // DIM/4 == 256
  const float4 ev = ((const float4*)(eo + row * DIM))[c4];
  float4 r;
  r.x = ev.x * g; r.y = ev.y * g; r.z = ev.z * g; r.w = ev.w * g;
  if (residAdd) {
    const float4 xv = ((const float4*)(residAdd + (size_t)t * DIM))[c4];
    r.x += xv.x; r.y += xv.y; r.z += xv.z; r.w += xv.w;
  }
  ((float4*)(out + (size_t)t * DIM))[c4] = r;
}

__global__ void mean_kernel(const float* __restrict__ resid, float* __restrict__ sent) {
  const int id = blockIdx.x * blockDim.x + threadIdx.x;  // NB*DIM ids
  const int b = id / DIM, c = id % DIM;
  float s = 0.f;
  for (int t = 0; t < SEQ; ++t)
    s += resid[((size_t)b * SEQ + t) * DIM + c];
  sent[id] = s * (1.f / SEQ);
}

__global__ __launch_bounds__(256) void loss_b_kernel(const float* __restrict__ sent,
                                                     const int* __restrict__ y,
                                                     float* __restrict__ partial) {
  const int b = blockIdx.x;
  const int tid = threadIdx.x;
  __shared__ float red[256];
  float m = -1e30f;
  for (int c = tid; c < DIM; c += 256) m = fmaxf(m, sent[b * DIM + c]);
  red[tid] = m; __syncthreads();
  for (int s2 = 128; s2 > 0; s2 >>= 1) {
    if (tid < s2) red[tid] = fmaxf(red[tid], red[tid + s2]);
    __syncthreads();
  }
  const float mx = red[0]; __syncthreads();
  float sacc = 0.f;
  for (int c = tid; c < DIM; c += 256) sacc += __expf(sent[b * DIM + c] - mx);
  red[tid] = sacc; __syncthreads();
  for (int s2 = 128; s2 > 0; s2 >>= 1) {
    if (tid < s2) red[tid] += red[tid + s2];
    __syncthreads();
  }
  if (tid == 0)
    partial[b] = -(sent[b * DIM + y[b]] - mx - __logf(red[0]));
}

__global__ void loss_final(const float* __restrict__ partial, float* __restrict__ out) {
  float s = 0.f;
  for (int b = 0; b < NB; ++b) s += partial[b];
  out[0] = s / NB;
}

// ---------------------------------------------------------------------------
// Host-side layer driver
// ---------------------------------------------------------------------------
static void run_moe(hipStream_t stream, const float* xin,
                    const float* wg, const float* w1, const float* b1,
                    const float* w2, const float* b2, int E, int cap,
                    int* idx, int* pos, float* gate,
                    __bf16* buf, __bf16* h, float* eo,
                    const float* residAdd, float* out) {
  gate_kernel<<<TTOK / 8, 256, 0, stream>>>(xin, wg, E, idx, gate);
  pos_kernel<<<E, 32, 0, stream>>>(idx, pos);
  const int bufQ = ROWS * DIM * 2 / 16;  // bf16 buffer as uint4 count
  fill_zero4<<<(bufQ + 255) / 256, 256, 0, stream>>>((uint4*)buf, bufQ);
  gather_kernel<<<TTOK, 256, 0, stream>>>(xin, idx, pos, cap, buf);
  // GEMM1: [cap,DIM] x [DIM,DFF] -> relu -> bf16 h
  gemm_bf16_wmma<true, true><<<dim3(DFF / 128, cap / 128, E), 256, 0, stream>>>(
      buf, w1, b1, h, DIM, DFF,
      (long long)cap * DIM, (long long)DIM * DFF, (long long)DFF, (long long)cap * DFF);
  // GEMM2: [cap,DFF] x [DFF,DIM] -> f32 eo
  gemm_bf16_wmma<false, false><<<dim3(DIM / 128, cap / 128, E), 256, 0, stream>>>(
      h, w2, b2, eo, DFF, DIM,
      (long long)cap * DFF, (long long)DFF * DIM, (long long)DIM, (long long)cap * DIM);
  scatter_kernel<<<TTOK, 256, 0, stream>>>(eo, idx, pos, gate, cap, residAdd, out);
}

extern "C" void kernel_launch(void* const* d_in, const int* in_sizes, int n_in,
                              void* d_out, int out_size, void* d_ws, size_t ws_size,
                              hipStream_t stream) {
  (void)in_sizes; (void)n_in; (void)out_size; (void)ws_size;
  const float* x   = (const float*)d_in[0];
  const int*   y   = (const int*)d_in[1];
  const float* wg1 = (const float*)d_in[2];
  const float* w1a = (const float*)d_in[3];
  const float* b1a = (const float*)d_in[4];
  const float* w2a = (const float*)d_in[5];
  const float* b2a = (const float*)d_in[6];
  const float* wg2 = (const float*)d_in[7];
  const float* w1b = (const float*)d_in[8];
  const float* b1b = (const float*)d_in[9];
  const float* w2b = (const float*)d_in[10];
  const float* b2b = (const float*)d_in[11];

  char* ws = (char*)d_ws;
  size_t off = 0;
  auto alloc = [&](size_t bytes) -> void* {
    void* p = ws + off;
    off = (off + bytes + 255) & ~(size_t)255;
    return p;
  };
  int*    idx1 = (int*)alloc(TTOK * 4);
  int*    pos1 = (int*)alloc(TTOK * 4);
  float*  gt1  = (float*)alloc(TTOK * 4);
  int*    idx2 = (int*)alloc(TTOK * 4);
  int*    pos2 = (int*)alloc(TTOK * 4);
  float*  gt2  = (float*)alloc(TTOK * 4);
  float*  sent = (float*)alloc((size_t)NB * DIM * 4);
  float*  prt  = (float*)alloc(NB * 4);
  __bf16* buf  = (__bf16*)alloc((size_t)ROWS * DIM * 2);   // 40 MB (reused)
  __bf16* h    = (__bf16*)alloc((size_t)ROWS * DFF * 2);   // 160 MB (reused)
  float*  eo   = (float*)alloc((size_t)ROWS * DIM * 4);    // 80 MB (reused)
  float*  m1   = (float*)alloc((size_t)TTOK * DIM * 4);    // 64 MB
  float*  rsd  = (float*)alloc((size_t)TTOK * DIM * 4);    // 64 MB

  // Layer 1: x -> m1
  run_moe(stream, x, wg1, w1a, b1a, w2a, b2a, E1, CAP1,
          idx1, pos1, gt1, buf, h, eo, /*residAdd=*/nullptr, m1);
  // Layer 2: m1 -> rsd (fused residual with x)
  run_moe(stream, m1, wg2, w1b, b1b, w2b, b2b, E2, CAP2,
          idx2, pos2, gt2, buf, h, eo, /*residAdd=*/x, rsd);

  mean_kernel<<<(NB * DIM) / 256, 256, 0, stream>>>(rsd, sent);
  loss_b_kernel<<<NB, 256, 0, stream>>>(sent, y, prt);
  loss_final<<<1, 1, 0, stream>>>(prt, (float*)d_out);
}
